// Decoder_27865747817086
// MI455X (gfx1250) — compile-verified, matrix operand order checked by reading, not run on hardware
//
#include <hip/hip_runtime.h>
#include <hip/hip_bf16.h>

#define BG   4096
#define LATD 128
#define MAXN 9
#define NEGS 0.01f

typedef _Float16 v16h __attribute__((ext_vector_type(16)));
typedef _Float16 v8h  __attribute__((ext_vector_type(8)));
typedef float    v8f  __attribute__((ext_vector_type(8)));

__device__ __forceinline__ float lrelu(float x){ return x > 0.f ? x : NEGS * x; }

__device__ __forceinline__ v16h make_frag(const _Float16* lo, const _Float16* hi){
  v8h a = *(const v8h*)lo;
  v8h b = *(const v8h*)hi;
  v16h f;
#pragma unroll
  for (int e = 0; e < 8; ++e){ f[e] = a[e]; f[e + 8] = b[e]; }
  return f;
}

__device__ __forceinline__ v8f wmma16(v16h a, v16h b, v8f c){
  return __builtin_amdgcn_wmma_f32_16x16x32_f16(false, a, false, b, (short)0, c, false, false);
}

// ---------------------------------------------------------------------------
// Pack a fp32 weight [Ksrc, 128] into f16 B-fragment layout [Kpad/32][128][2][16]
// element (kc,n,hi,e): K = kc*32 + hi*8 + (e<8 ? e : e+8)
// ---------------------------------------------------------------------------
__global__ __launch_bounds__(256)
void k_pack(const float* __restrict__ W, _Float16* __restrict__ out, int Ksrc, int Kpad){
  const int total = (Kpad / 32) * 128 * 2 * 16;
  for (int idx = blockIdx.x * blockDim.x + threadIdx.x; idx < total; idx += gridDim.x * blockDim.x){
    const int e  = idx & 15;
    const int hi = (idx >> 4) & 1;
    const int n  = (idx >> 5) & 127;
    const int kc = idx >> 12;
    const int K  = kc * 32 + hi * 8 + (e < 8 ? e : e + 8);
    const float v = (K < Ksrc) ? W[K * 128 + n] : 0.f;
    out[idx] = (_Float16)v;
  }
}

// ---------------------------------------------------------------------------
// Front end: pred_num, glob MLP, and pre-aggregated GIN layer-0 input
// s0[i] = x_i + sum_{j<n} x_j  (A = all-ones over valid nodes incl. self loop)
// stored padded [16,96] f16 per graph.
// ---------------------------------------------------------------------------
__global__ __launch_bounds__(128)
void k_frontend(const float* __restrict__ gv, const int* __restrict__ nn,
                const float* __restrict__ lap,
                const float* Wn1, const float* bn1, const float* Wn2, const float* bn2,
                const float* Wn3, const float* bn3,
                const float* Wg1, const float* bg1, const float* Wg2, const float* bg2,
                const float* Wg3, const float* bg3,
                _Float16* __restrict__ s0, float* __restrict__ pred)
{
  __shared__ float xs[LATD], p1[LATD], p2[LATD], red[LATD];
  __shared__ float g1s[64], g2s[64], g3s[64];
  __shared__ float lapS[81], lsum[MAXN];
  const int b = blockIdx.x, t = threadIdx.x;
  const int n = nn[b];

  xs[t] = gv[b * LATD + t];
  if (t < 81) lapS[t] = lap[(n - 1) * 81 + t];
  __syncthreads();

  if (t < MAXN){
    float s = 0.f;
    for (int j = 0; j < n; ++j) s += lapS[j * MAXN + t];
    lsum[t] = s;
  }

  // pred_num net (ReLU)
  float a = 0.f;
  for (int k = 0; k < LATD; ++k) a += xs[k] * Wn1[k * LATD + t];
  a += bn1[t]; p1[t] = a > 0.f ? a : 0.f;
  __syncthreads();
  a = 0.f;
  for (int k = 0; k < LATD; ++k) a += p1[k] * Wn2[k * LATD + t];
  a += bn2[t]; p2[t] = a > 0.f ? a : 0.f;
  __syncthreads();
  red[t] = p2[t] * Wn3[t];
  __syncthreads();
  for (int s = 64; s > 0; s >>= 1){ if (t < s) red[t] += red[t + s]; __syncthreads(); }
  if (t == 0) pred[b] = red[0] + bn3[0];

  // glob MLP (leaky relu on first two layers)
  if (t < 64){
    float a1 = 0.f;
    for (int k = 0; k < LATD; ++k) a1 += xs[k] * Wg1[k * 64 + t];
    g1s[t] = lrelu(a1 + bg1[t]);
  }
  __syncthreads();
  if (t < 64){
    float a2 = 0.f;
    for (int k = 0; k < 64; ++k) a2 += g1s[k] * Wg2[k * 64 + t];
    g2s[t] = lrelu(a2 + bg2[t]);
  }
  __syncthreads();
  if (t < 64){
    float a3 = 0.f;
    for (int k = 0; k < 64; ++k) a3 += g2s[k] * Wg3[k * 64 + t];
    g3s[t] = a3 + bg3[t];
  }
  __syncthreads();

  _Float16* outp = s0 + (size_t)b * 16 * 96;
  for (int idx = t; idx < 16 * 96; idx += LATD){
    const int row = idx / 96, c = idx % 96;
    float v = 0.f;
    if (row < n){
      if (c < 64)       v = (1.f + (float)n) * g3s[c];
      else if (c < 73)  v = lapS[row * MAXN + (c - 64)] + lsum[c - 64];
    }
    outp[idx] = (_Float16)v;
  }
}

// ---------------------------------------------------------------------------
// Fused 3-layer GIN. One block per graph, 8 waves, each wave owns 16 N-cols.
// ---------------------------------------------------------------------------
struct GinArgs {
  const _Float16* wa[3];
  const _Float16* wb[3];
  const float* ba[3];
  const float* bb[3];
  const float* gamma[2];
  const float* beta[2];
  const float* alpha[2];
};

__global__ __launch_bounds__(256)
void k_gin(const int* __restrict__ nn, GinArgs args,
           _Float16* __restrict__ hOut, const _Float16* __restrict__ s0)
{
  __shared__ __align__(16) _Float16 sBuf[16][128];
  __shared__ __align__(16) _Float16 tBuf[16][128];
  const int b    = blockIdx.x;
  const int lane = threadIdx.x & 31;
  const int wv   = threadIdx.x >> 5;
  const int n    = nn[b];
  const float inv_n = 1.f / (float)n;
  const int m    = lane & 15;
  const int kb   = (lane >= 16) ? 8 : 0;
  const int hi   = kb >> 3;
  const int nIdx = wv * 16 + (lane & 15);
  const int roff = (lane < 16) ? 0 : 8;

  for (int l = 0; l < 3; ++l){
    // ---- GEMM1: s @ Wa, + bias, leaky relu -> tBuf --------------------
    v8f acc = {0.f,0.f,0.f,0.f,0.f,0.f,0.f,0.f};
    const _Float16* wa = args.wa[l];
    if (l == 0){
      const _Float16* arow = s0 + ((size_t)b * 16 + m) * 96;
#pragma unroll
      for (int kc = 0; kc < 3; ++kc){
        v16h af = make_frag(arow + kc * 32 + kb, arow + kc * 32 + kb + 16);
        v16h bf = *(const v16h*)(wa + ((kc * 128 + nIdx) * 2 + hi) * 16);
        acc = wmma16(af, bf, acc);
      }
    } else {
#pragma unroll
      for (int kc = 0; kc < 4; ++kc){
        v16h af = make_frag(&sBuf[m][kc * 32 + kb], &sBuf[m][kc * 32 + kb + 16]);
        v16h bf = *(const v16h*)(wa + ((kc * 128 + nIdx) * 2 + hi) * 16);
        acc = wmma16(af, bf, acc);
      }
    }
    const float bav = args.ba[l][nIdx];
#pragma unroll
    for (int v = 0; v < 8; ++v)
      tBuf[roff + v][nIdx] = (_Float16)lrelu(acc[v] + bav);
    __syncthreads();

    // ---- GEMM2: t1 @ Wb + bias ---------------------------------------
    v8f acc2 = {0.f,0.f,0.f,0.f,0.f,0.f,0.f,0.f};
    const _Float16* wb = args.wb[l];
#pragma unroll
    for (int kc = 0; kc < 4; ++kc){
      v16h af = make_frag(&tBuf[m][kc * 32 + kb], &tBuf[m][kc * 32 + kb + 16]);
      v16h bf = *(const v16h*)(wb + ((kc * 128 + nIdx) * 2 + hi) * 16);
      acc2 = wmma16(af, bf, acc2);
    }
    float hv[8];
    const float bbv = args.bb[l][nIdx];
#pragma unroll
    for (int v = 0; v < 8; ++v) hv[v] = acc2[v] + bbv;

    if (l < 2){
      // GraphNorm over valid nodes: column stats are intra-wave
      float s = 0.f;
#pragma unroll
      for (int v = 0; v < 8; ++v) if (roff + v < n) s += hv[v];
      s += __shfl_xor(s, 16, 32);
      const float mean = s * inv_n;
      const float al = args.alpha[l][nIdx];
      float cv[8]; float ss = 0.f;
#pragma unroll
      for (int v = 0; v < 8; ++v){
        const float c = (roff + v < n) ? (hv[v] - al * mean) : 0.f;
        cv[v] = c; ss += c * c;
      }
      ss += __shfl_xor(ss, 16, 32);
      const float rinv = rsqrtf(ss * inv_n + 1e-5f);
      const float ga = args.gamma[l][nIdx], be = args.beta[l][nIdx];
#pragma unroll
      for (int v = 0; v < 8; ++v)
        hv[v] = (roff + v < n) ? lrelu(ga * cv[v] * rinv + be) : 0.f;
      // pre-aggregate for next layer: s_i = h_i + colsum(h) over valid nodes
      float cs = 0.f;
#pragma unroll
      for (int v = 0; v < 8; ++v) cs += hv[v];
      cs += __shfl_xor(cs, 16, 32);
      __syncthreads();
#pragma unroll
      for (int v = 0; v < 8; ++v)
        sBuf[roff + v][nIdx] = (_Float16)((roff + v < n) ? hv[v] + cs : 0.f);
      __syncthreads();
    } else {
      _Float16* orow = hOut + (size_t)b * 16 * 128;
#pragma unroll
      for (int v = 0; v < 8; ++v)
        orow[(roff + v) * 128 + nIdx] = (_Float16)((roff + v < n) ? hv[v] : 0.f);
    }
  }
}

// ---------------------------------------------------------------------------
// Edge head (mode 0) / node head (mode 1). One block = 16 rows x N=128.
// ---------------------------------------------------------------------------
__global__ __launch_bounds__(256)
void k_head(const _Float16* __restrict__ H, const int* __restrict__ nn,
            const _Float16* __restrict__ w1P, const float* __restrict__ b1,
            const _Float16* __restrict__ w2P, const float* __restrict__ b2,
            const float* __restrict__ w3, const float* __restrict__ b3,
            float* __restrict__ out, int mode)
{
  __shared__ __align__(16) _Float16 tBuf[16][128];
  __shared__ __align__(16) _Float16 uBuf[16][128];
  const int lane = threadIdx.x & 31;
  const int wv   = threadIdx.x >> 5;
  const int m    = lane & 15;
  const int kb   = (lane >= 16) ? 8 : 0;
  const int hi   = kb >> 3;
  const int nIdx = wv * 16 + (lane & 15);
  const int roff = (lane < 16) ? 0 : 8;
  const long r0  = (long)blockIdx.x * 16;

  const long r = r0 + m;
  const _Float16 *pa, *pb;
  if (mode == 0){
    const long gb = r / 81; const int e = (int)(r % 81);
    pa = H + (gb * 16 + e / 9) * 128;
    pb = H + (gb * 16 + e % 9) * 128;
  } else {
    const long gb = r / 9; const int i = (int)(r % 9);
    pa = H + (gb * 16 + i) * 128;
    pb = pa;
  }

  v8f acc = {0.f,0.f,0.f,0.f,0.f,0.f,0.f,0.f};
#pragma unroll
  for (int kc = 0; kc < 4; ++kc){
    const int ko = kc * 32 + kb;
    __builtin_prefetch(w1P + ((((kc + 1) & 3) * 128 + nIdx) * 2 + hi) * 16, 0, 3);
    v16h af;
    if (mode == 0){
      v8h la = *(const v8h*)(pa + ko),      lb = *(const v8h*)(pb + ko);
      v8h ha = *(const v8h*)(pa + ko + 16), hb = *(const v8h*)(pb + ko + 16);
#pragma unroll
      for (int e2 = 0; e2 < 8; ++e2){
        af[e2]     = (_Float16)(((float)la[e2] + (float)lb[e2]) * 0.5f);
        af[e2 + 8] = (_Float16)(((float)ha[e2] + (float)hb[e2]) * 0.5f);
      }
    } else {
      af = make_frag(pa + ko, pa + ko + 16);
    }
    v16h bf = *(const v16h*)(w1P + ((kc * 128 + nIdx) * 2 + hi) * 16);
    acc = wmma16(af, bf, acc);
  }
  const float b1v = b1[nIdx];
#pragma unroll
  for (int v = 0; v < 8; ++v) tBuf[roff + v][nIdx] = (_Float16)lrelu(acc[v] + b1v);
  __syncthreads();

  v8f acc2 = {0.f,0.f,0.f,0.f,0.f,0.f,0.f,0.f};
#pragma unroll
  for (int kc = 0; kc < 4; ++kc){
    v16h af = make_frag(&tBuf[m][kc * 32 + kb], &tBuf[m][kc * 32 + kb + 16]);
    v16h bf = *(const v16h*)(w2P + ((kc * 128 + nIdx) * 2 + hi) * 16);
    acc2 = wmma16(af, bf, acc2);
  }
  const float b2v = b2[nIdx];
#pragma unroll
  for (int v = 0; v < 8; ++v) uBuf[roff + v][nIdx] = (_Float16)lrelu(acc2[v] + b2v);
  __syncthreads();

  const int NO = (mode == 0) ? 5 : 4;
  const int tid = threadIdx.x;
  if (tid < 16 * NO){
    const int row = tid / NO, c = tid % NO;
    const long rr = r0 + row;
    float a = b3[c];
    for (int k = 0; k < 128; ++k) a += (float)uBuf[row][k] * w3[k * NO + c];
    if (mode == 0){
      const long gb = rr / 81; const int e = (int)(rr % 81);
      const int i = e / 9, j = e % 9, n = nn[gb];
      out[(gb * 81 + e) * 5 + c] = (i < n && j < n) ? a : 0.f;
    } else {
      const long gb = rr / 9; const int i = (int)(rr % 9);
      const int n = nn[gb];
      out[(gb * 9 + i) * 4 + c] = (i < n) ? a : 0.f;
    }
  }
}

// ---------------------------------------------------------------------------
extern "C" void kernel_launch(void* const* d_in, const int* in_sizes, int n_in,
                              void* d_out, int out_size, void* d_ws, size_t ws_size,
                              hipStream_t stream)
{
  (void)in_sizes; (void)n_in; (void)out_size; (void)ws_size;

  const float* gv  = (const float*)d_in[0];
  const int*   nn  = (const int*)d_in[1];
  const float* lap = (const float*)d_in[2];
  // num_net
  const float *Wn1=(const float*)d_in[3],  *bn1=(const float*)d_in[4];
  const float *Wn2=(const float*)d_in[5],  *bn2=(const float*)d_in[6];
  const float *Wn3=(const float*)d_in[7],  *bn3=(const float*)d_in[8];
  // glob
  const float *Wg1=(const float*)d_in[9],  *bg1=(const float*)d_in[10];
  const float *Wg2=(const float*)d_in[11], *bg2=(const float*)d_in[12];
  const float *Wg3=(const float*)d_in[13], *bg3=(const float*)d_in[14];
  // gin layers: {A:(W,b), B:(W,b)} x3
  const float *Wa0=(const float*)d_in[15], *ba0=(const float*)d_in[16];
  const float *Wb0=(const float*)d_in[17], *bb0=(const float*)d_in[18];
  const float *Wa1=(const float*)d_in[19], *ba1=(const float*)d_in[20];
  const float *Wb1=(const float*)d_in[21], *bb1=(const float*)d_in[22];
  const float *Wa2=(const float*)d_in[23], *ba2=(const float*)d_in[24];
  const float *Wb2=(const float*)d_in[25], *bb2=(const float*)d_in[26];
  // graph norms: gamma, beta, alpha x2
  const float *g0g=(const float*)d_in[27], *g0b=(const float*)d_in[28], *g0a=(const float*)d_in[29];
  const float *g1g=(const float*)d_in[30], *g1b=(const float*)d_in[31], *g1a=(const float*)d_in[32];
  // feat
  const float *Wf1=(const float*)d_in[33], *bf1=(const float*)d_in[34];
  const float *Wf2=(const float*)d_in[35], *bf2=(const float*)d_in[36];
  const float *Wf3=(const float*)d_in[37], *bf3=(const float*)d_in[38];
  // edge
  const float *We1=(const float*)d_in[39], *be1=(const float*)d_in[40];
  const float *We2=(const float*)d_in[41], *be2=(const float*)d_in[42];
  const float *We3=(const float*)d_in[43], *be3=(const float*)d_in[44];

  _Float16* ws = (_Float16*)d_ws;
  // packed weight layout (offsets in halfs)
  const size_t WA0 = 0;                     // 96x128  -> 12288
  const size_t WB0 = WA0 + 12288;
  const size_t WA1 = WB0 + 16384;
  const size_t WB1 = WA1 + 16384;
  const size_t WA2 = WB1 + 16384;
  const size_t WB2 = WA2 + 16384;
  const size_t WF1 = WB2 + 16384;
  const size_t WF2 = WF1 + 16384;
  const size_t WE1 = WF2 + 16384;
  const size_t WE2 = WE1 + 16384;
  const size_t S0  = WE2 + 16384;           // [B,16,96] f16
  const size_t HF  = S0 + (size_t)BG * 16 * 96;  // [B,16,128] f16

  k_pack<<<64, 256, 0, stream>>>(Wa0, ws + WA0, 73, 96);
  k_pack<<<64, 256, 0, stream>>>(Wb0, ws + WB0, 128, 128);
  k_pack<<<64, 256, 0, stream>>>(Wa1, ws + WA1, 128, 128);
  k_pack<<<64, 256, 0, stream>>>(Wb1, ws + WB1, 128, 128);
  k_pack<<<64, 256, 0, stream>>>(Wa2, ws + WA2, 128, 128);
  k_pack<<<64, 256, 0, stream>>>(Wb2, ws + WB2, 128, 128);
  k_pack<<<64, 256, 0, stream>>>(Wf1, ws + WF1, 128, 128);
  k_pack<<<64, 256, 0, stream>>>(Wf2, ws + WF2, 128, 128);
  k_pack<<<64, 256, 0, stream>>>(We1, ws + WE1, 128, 128);
  k_pack<<<64, 256, 0, stream>>>(We2, ws + WE2, 128, 128);

  float* nodeOut = (float*)d_out;                       // [B,9,4]
  float* edgeOut = nodeOut + (size_t)BG * 9 * 4;        // [B,9,9,5]
  float* predOut = edgeOut + (size_t)BG * 81 * 5;       // [B]

  k_frontend<<<BG, 128, 0, stream>>>(gv, nn, lap,
      Wn1, bn1, Wn2, bn2, Wn3, bn3,
      Wg1, bg1, Wg2, bg2, Wg3, bg3,
      ws + S0, predOut);

  GinArgs ga;
  ga.wa[0] = ws + WA0; ga.wa[1] = ws + WA1; ga.wa[2] = ws + WA2;
  ga.wb[0] = ws + WB0; ga.wb[1] = ws + WB1; ga.wb[2] = ws + WB2;
  ga.ba[0] = ba0; ga.ba[1] = ba1; ga.ba[2] = ba2;
  ga.bb[0] = bb0; ga.bb[1] = bb1; ga.bb[2] = bb2;
  ga.gamma[0] = g0g; ga.gamma[1] = g1g;
  ga.beta[0]  = g0b; ga.beta[1]  = g1b;
  ga.alpha[0] = g0a; ga.alpha[1] = g1a;
  k_gin<<<BG, 256, 0, stream>>>(nn, ga, ws + HF, ws + S0);

  // edges: B*81 rows / 16 per block = 20736 blocks
  k_head<<<(BG * 81) / 16, 256, 0, stream>>>(ws + HF, nn,
      ws + WE1, be1, ws + WE2, be2, We3, be3, edgeOut, 0);
  // nodes: B*9 rows / 16 per block = 2304 blocks
  k_head<<<(BG * 9) / 16, 256, 0, stream>>>(ws + HF, nn,
      ws + WF1, bf1, ws + WF2, bf2, Wf3, bf3, nodeOut, 1);
}
